// TransLayer2_85899345920258
// MI455X (gfx1250) — matrix operand order repaired and checked by reading, not compiled, not run on hardware
//
#include <hip/hip_runtime.h>
#include <math.h>

// ---------------------------------------------------------------------------
// MI455X / gfx1250: all GEMMs via v_wmma_f32_16x16x32_f16 with fragments
// built directly from global memory (no LDS round trip, no divergent guards),
// fused per-(b,n,h) time-axis attention, warp-reduction LayerNorm (wave32).
// ---------------------------------------------------------------------------

typedef __attribute__((ext_vector_type(16))) _Float16 v16h;
typedef __attribute__((ext_vector_type(8)))  float    v8f;

namespace {
constexpr int Bc = 16, Lc = 48, Nc = 207, Dc = 64, DFFc = 256, Hc = 8, Ec = 8;
constexpr int CINc = 5 * Dc;                               // 320
constexpr long long Tok   = (long long)Bc * Lc * Nc;       // 158976 tokens
constexpr long long SZ_D  = Tok * Dc;                      // 10,174,464
constexpr long long SZ_NN = Tok * Nc;                      // 32,908,032
}

static __device__ __forceinline__ int imin(int a, int b) { return a < b ? a : b; }

// ---------------------------------------------------------------------------
// Generic strided, batched GEMM:  C = act(A * B + bias)
//   A(m,k) = A[(batch % Abmod)*Ab + m*lda + k]           (k-stride 1)
//   B(k,n) = B[batch*Bb + k*sbk + n*sbn]
//   C(m,n) = C[batch*Cb + m*scm + n*scn]
//   bias_mode: 0 none, 1 per-column, 2 per-row; offset (batch % bmod)*bstride
// Block = 256 threads = 8 waves. Each wave owns a 16(M) x 64(N) strip:
// one A fragment is reused across four B fragments -> 4 WMMAs per K-step.
// Fragments are gathered straight from global memory with clamped addresses;
// only the K-tail (wave-uniform branch) masks the A fragment to zero.
// sbk==1 (all weight GEMMs) takes a contiguous path that the compiler merges
// into global_load_b128 clauses, matching the A path.
// ---------------------------------------------------------------------------
__global__ void __launch_bounds__(256)
gemm_f16_wmma(const float* __restrict__ A, unsigned long long Ab, unsigned Abmod, int lda,
              const float* __restrict__ Bm, unsigned long long Bb, long long sbk, long long sbn,
              float* __restrict__ C, unsigned long long Cb, long long scm, long long scn,
              const float* __restrict__ bias, int bias_mode, unsigned bmod, int bstride,
              int M, int N, int K, int relu)
{
    const int lane = threadIdx.x & 31;
    const int wave = threadIdx.x >> 5;
    const int m0 = (blockIdx.y * 8 + wave) * 16;
    if (m0 >= M) return;                                  // wave-uniform
    const int n0 = blockIdx.x * 64;
    const unsigned batch = blockIdx.z;

    const float* Ap = A + (unsigned long long)(batch % Abmod) * Ab;
    const float* Bp = Bm + (unsigned long long)batch * Bb;
    float*       Cp = C + (unsigned long long)batch * Cb;

    // A fragment geometry: lane m = lane&15, K half 0-15 / 16-31
    const int am  = lane & 15;
    const int akb = (lane < 16) ? 0 : 16;
    const float* Arow = Ap + (long long)imin(m0 + am, M - 1) * lda;

    // B fragment geometry: lane n = lane&15, K half 0-15 / 16-31
    const int bn  = lane & 15;
    const int bkb = (lane < 16) ? 0 : 16;
    const float* Bcol0 = Bp + (long long)imin(n0 + 0 * 16 + bn, N - 1) * sbn;
    const float* Bcol1 = Bp + (long long)imin(n0 + 1 * 16 + bn, N - 1) * sbn;
    const float* Bcol2 = Bp + (long long)imin(n0 + 2 * 16 + bn, N - 1) * sbn;
    const float* Bcol3 = Bp + (long long)imin(n0 + 3 * 16 + bn, N - 1) * sbn;

    const bool contig = (sbk == 1);                        // wave-uniform

    v8f acc[4] = {};
    for (int k0 = 0; k0 < K; k0 += 32) {
        __builtin_prefetch(Arow + k0 + 32, 0, 1);         // global_prefetch_b8
        v16h af;
        v16h bf0, bf1, bf2, bf3;
        if (k0 + 32 <= K) {                               // fast path: no masking
            #pragma unroll
            for (int j = 0; j < 8; ++j) {
                af[j]     = (_Float16)Arow[k0 + akb + j];
                af[8 + j] = (_Float16)Arow[k0 + akb + 8 + j];
            }
            if (contig) {                                 // unit k-stride: merges to b128
                const float* c0 = Bcol0 + k0 + bkb;
                const float* c1 = Bcol1 + k0 + bkb;
                const float* c2 = Bcol2 + k0 + bkb;
                const float* c3 = Bcol3 + k0 + bkb;
                #pragma unroll
                for (int j = 0; j < 16; ++j) {
                    bf0[j] = (_Float16)c0[j];
                    bf1[j] = (_Float16)c1[j];
                    bf2[j] = (_Float16)c2[j];
                    bf3[j] = (_Float16)c3[j];
                }
            } else {                                      // strided gather (support/As/h)
                #pragma unroll
                for (int j = 0; j < 16; ++j) {
                    long long ko = (long long)(k0 + bkb + j) * sbk;
                    bf0[j] = (_Float16)Bcol0[ko];
                    bf1[j] = (_Float16)Bcol1[ko];
                    bf2[j] = (_Float16)Bcol2[ko];
                    bf3[j] = (_Float16)Bcol3[ko];
                }
            }
        } else {                                          // K tail: clamp + zero A only
            #pragma unroll
            for (int j = 0; j < 8; ++j) {
                int gk0 = k0 + akb + j, gk1 = k0 + akb + 8 + j;
                float a0 = Arow[imin(gk0, K - 1)];
                float a1 = Arow[imin(gk1, K - 1)];
                a0 = (gk0 < K) ? a0 : 0.f;
                a1 = (gk1 < K) ? a1 : 0.f;
                af[j] = (_Float16)a0;  af[8 + j] = (_Float16)a1;
            }
            #pragma unroll
            for (int j = 0; j < 16; ++j) {
                long long ko = (long long)imin(k0 + bkb + j, K - 1) * sbk;
                bf0[j] = (_Float16)Bcol0[ko];
                bf1[j] = (_Float16)Bcol1[ko];
                bf2[j] = (_Float16)Bcol2[ko];
                bf3[j] = (_Float16)Bcol3[ko];
            }
        }
        acc[0] = __builtin_amdgcn_wmma_f32_16x16x32_f16(false, af, false, bf0, (short)0, acc[0], false, false);
        acc[1] = __builtin_amdgcn_wmma_f32_16x16x32_f16(false, af, false, bf1, (short)0, acc[1], false, false);
        acc[2] = __builtin_amdgcn_wmma_f32_16x16x32_f16(false, af, false, bf2, (short)0, acc[2], false, false);
        acc[3] = __builtin_amdgcn_wmma_f32_16x16x32_f16(false, af, false, bf3, (short)0, acc[3], false, false);
    }

    // store: VGPR r holds M = r + (lane<16 ? 0 : 8), N = lane&15
    const int mh = (lane < 16) ? 0 : 8;
    const unsigned bb = batch % bmod;
    #pragma unroll
    for (int t = 0; t < 4; ++t) {
        int gn = n0 + t * 16 + bn;
        if (gn < N) {
            #pragma unroll
            for (int r = 0; r < 8; ++r) {
                int gm = m0 + mh + r;
                if (gm < M) {
                    float vv = acc[t][r];
                    if (bias_mode == 1)      vv += bias[(long long)bb * bstride + gn];
                    else if (bias_mode == 2) vv += bias[(long long)bb * bstride + gm];
                    if (relu) vv = vv > 0.f ? vv : 0.f;
                    Cp[(long long)gm * scm + (long long)gn * scn] = vv;
                }
            }
        }
    }
}

// ---------------------------------------------------------------------------
// Fused time-axis attention for one (b, n, h): scores = QK^T/sqrt(E) (3x3
// WMMA tiles, E=8 zero-padded to K=32), softmax over L=48 in LDS, then P*V
// (3 tiles x 2 K-steps, L padded 48->64, E padded 8->16).
// ---------------------------------------------------------------------------
__global__ void __launch_bounds__(32)
attn_time_wmma(const float* __restrict__ Q, const float* __restrict__ Kp,
               const float* __restrict__ V, float* __restrict__ O)
{
    __shared__ _Float16 Qs[48 * 32];
    __shared__ _Float16 Ks[48 * 32];
    __shared__ _Float16 Vs[64 * 16];
    __shared__ float    S [48 * 48];
    __shared__ _Float16 Ps[48 * 64];

    const int lane = threadIdx.x;
    const int bid  = blockIdx.x;                 // (b*N + n)*H + h
    const int hh   = bid % Hc;
    const int nn   = (bid / Hc) % Nc;
    const int bb   = bid / (Hc * Nc);
    const long long ND   = (long long)Nc * Dc;   // time-step stride
    const long long base = (((long long)bb * Lc) * Nc + nn) * Dc + hh * Ec;

    // zero-fill padded tiles, then overwrite valid region (same-wave LDS ops
    // are in order; unconditional global loads batch under one wait)
    for (int i = lane; i < 48 * 32; i += 32) { Qs[i] = (_Float16)0.f; Ks[i] = (_Float16)0.f; }
    for (int i = lane; i < 64 * 16; i += 32) Vs[i] = (_Float16)0.f;
    for (int idx = lane; idx < 48 * 8; idx += 32) {
        int l = idx >> 3, e = idx & 7;
        long long g = base + l * ND + e;
        Qs[l * 32 + e] = (_Float16)Q[g];
        Ks[l * 32 + e] = (_Float16)Kp[g];
        Vs[l * 16 + e] = (_Float16)V[g];
    }
    __syncthreads();

    const float scale = 0.35355339059327373f;    // 1/sqrt(8)
    const int am = lane & 15, akb = (lane < 16) ? 0 : 16;
    const int bn = lane & 15, bkb = (lane < 16) ? 0 : 16;
    const int cn = lane & 15, mh  = (lane < 16) ? 0 : 8;

    #pragma unroll
    for (int ti = 0; ti < 3; ++ti) {
        #pragma unroll
        for (int tj = 0; tj < 3; ++tj) {
            v16h af, bf;
            #pragma unroll
            for (int j = 0; j < 8; ++j) {
                af[j]     = Qs[(ti * 16 + am) * 32 + akb + j];
                af[8 + j] = Qs[(ti * 16 + am) * 32 + akb + 8 + j];
            }
            #pragma unroll
            for (int j = 0; j < 16; ++j) bf[j] = Ks[(tj * 16 + bn) * 32 + bkb + j];
            v8f c = {};
            c = __builtin_amdgcn_wmma_f32_16x16x32_f16(false, af, false, bf,
                                                       (short)0, c, false, false);
            #pragma unroll
            for (int r = 0; r < 8; ++r)
                S[(ti * 16 + mh + r) * 48 + tj * 16 + cn] = c[r] * scale;
        }
    }
    __syncthreads();

    // softmax over each row of S, write probabilities (K-padded to 64) as f16
    for (int r = lane; r < 48; r += 32) {
        float mx = -1e30f;
        for (int s = 0; s < 48; ++s) mx = fmaxf(mx, S[r * 48 + s]);
        float sum = 0.f;
        for (int s = 0; s < 48; ++s) { float e = __expf(S[r * 48 + s] - mx); S[r * 48 + s] = e; sum += e; }
        float inv = 1.f / sum;
        for (int s = 0; s < 64; ++s)
            Ps[r * 64 + s] = (s < 48) ? (_Float16)(S[r * 48 + s] * inv) : (_Float16)0.f;
    }
    __syncthreads();

    #pragma unroll
    for (int ti = 0; ti < 3; ++ti) {
        v8f c = {};
        #pragma unroll
        for (int ks = 0; ks < 2; ++ks) {
            v16h af, bf;
            #pragma unroll
            for (int j = 0; j < 8; ++j) {
                af[j]     = Ps[(ti * 16 + am) * 64 + ks * 32 + akb + j];
                af[8 + j] = Ps[(ti * 16 + am) * 64 + ks * 32 + akb + 8 + j];
            }
            #pragma unroll
            for (int j = 0; j < 16; ++j) bf[j] = Vs[(ks * 32 + bkb + j) * 16 + bn];
            c = __builtin_amdgcn_wmma_f32_16x16x32_f16(false, af, false, bf,
                                                       (short)0, c, false, false);
        }
        if (cn < Ec) {
            #pragma unroll
            for (int r = 0; r < 8; ++r) {
                int l = ti * 16 + mh + r;
                O[base + l * ND + cn] = c[r];
            }
        }
    }
}

// ---------------------------------------------------------------------------
// out[t,:] = LayerNorm(a[t,:] + b[t,:]) * g + be       (D=64, one wave/token)
// ---------------------------------------------------------------------------
__global__ void __launch_bounds__(256)
ln_add_kernel(const float* __restrict__ A, const float* __restrict__ Bq,
              const float* __restrict__ g, const float* __restrict__ be,
              float* __restrict__ out, long long ntok)
{
    const int wave = threadIdx.x >> 5, lane = threadIdx.x & 31;
    const long long t = (long long)blockIdx.x * 8 + wave;
    if (t >= ntok) return;
    const float* a = A + t * Dc;
    const float* b = Bq + t * Dc;
    float v0 = a[lane] + b[lane];
    float v1 = a[lane + 32] + b[lane + 32];
    float s = v0 + v1;
    #pragma unroll
    for (int o = 16; o > 0; o >>= 1) s += __shfl_xor(s, o, 32);
    const float mu = s * (1.f / 64.f);
    float d0 = v0 - mu, d1 = v1 - mu;
    float q = d0 * d0 + d1 * d1;
    #pragma unroll
    for (int o = 16; o > 0; o >>= 1) q += __shfl_xor(q, o, 32);
    const float inv = rsqrtf(q * (1.f / 64.f) + 1e-5f);
    out[t * Dc + lane]      = d0 * inv * g[lane]      + be[lane];
    out[t * Dc + lane + 32] = d1 * inv * g[lane + 32] + be[lane + 32];
}

// h[b,l,d,n] = data[b,l,n,d]   (slice 0 of the 5*D concat buffer)
__global__ void __launch_bounds__(256)
transpose_x0(const float* __restrict__ data, float* __restrict__ hbuf, long long total)
{
    long long i = (long long)blockIdx.x * 256 + threadIdx.x;
    if (i >= total) return;
    int d = (int)(i & 63);
    long long r = i >> 6;
    int n = (int)(r % Nc);
    long long bl = r / Nc;
    hbuf[(bl * CINc + d) * Nc + n] = data[i];
}

// ---------------------------------------------------------------------------
extern "C" void kernel_launch(void* const* d_in, const int* in_sizes, int n_in,
                              void* d_out, int out_size, void* d_ws, size_t ws_size,
                              hipStream_t stream) {
    (void)in_sizes; (void)n_in; (void)out_size; (void)ws_size;
    const float* x       = (const float*)d_in[0];
    const float* memory  = (const float*)d_in[1];
    const float* data    = (const float*)d_in[2];
    const float* support = (const float*)d_in[3];
    const float* Wq1 = (const float*)d_in[4],  *Wk1 = (const float*)d_in[5];
    const float* Wv1 = (const float*)d_in[6],  *Wo1 = (const float*)d_in[7];
    const float* Wq2 = (const float*)d_in[8],  *Wk2 = (const float*)d_in[9];
    const float* Wv2 = (const float*)d_in[10], *Wo2 = (const float*)d_in[11];
    const float* bq1 = (const float*)d_in[12], *bk1 = (const float*)d_in[13];
    const float* bv1 = (const float*)d_in[14], *bo1 = (const float*)d_in[15];
    const float* bq2 = (const float*)d_in[16], *bk2 = (const float*)d_in[17];
    const float* bv2 = (const float*)d_in[18], *bo2 = (const float*)d_in[19];
    const float* W1  = (const float*)d_in[20], *b1  = (const float*)d_in[21];
    const float* W2  = (const float*)d_in[22], *b2  = (const float*)d_in[23];
    const float* W3  = (const float*)d_in[24], *b3  = (const float*)d_in[25];
    const float* W4  = (const float*)d_in[26], *b4  = (const float*)d_in[27];
    const float* WN2 = (const float*)d_in[28], *bN2 = (const float*)d_in[29];
    const float* gx1 = (const float*)d_in[30], *gx2 = (const float*)d_in[31];
    const float* gm1 = (const float*)d_in[32], *gm2 = (const float*)d_in[33];
    const float* bx1 = (const float*)d_in[34], *bx2 = (const float*)d_in[35];
    const float* bm1 = (const float*)d_in[36], *bm2 = (const float*)d_in[37];
    const float* Wg  = (const float*)d_in[38], *bg  = (const float*)d_in[39];

    float* ws     = (float*)d_ws;
    float* qb     = ws;                  // [T,D]
    float* kb     = ws + SZ_D;
    float* vb     = ws + 2 * SZ_D;
    float* attn   = ws + 3 * SZ_D;
    float* proj   = ws + 4 * SZ_D;       // later reused as y
    float* xln    = ws + 5 * SZ_D;
    float* memln  = ws + 6 * SZ_D;
    float* mbuf   = ws + 7 * SZ_D;
    float* ffbuf  = ws;                  // [T,DFF] reuses q/k/v/attn after attention phase
    float* ybuf   = ws + 4 * SZ_D;
    float* Asbuf  = ws + 8 * SZ_D;                 // [B,L,N,N]
    float* hbuf   = ws + 8 * SZ_D + SZ_NN;         // [B,L,5D,N]
    float* out0 = (float*)d_out;
    float* out1 = out0 + SZ_D;
    float* out2 = out0 + 2 * SZ_D;

    const int T = (int)Tok;
    const unsigned BIG = 1u << 30;

    auto gemm = [&](const float* A, unsigned long long Ab, unsigned Abmod, int lda,
                    const float* Bm, unsigned long long Bb, long long sbk, long long sbn,
                    float* C, unsigned long long Cb, long long scm, long long scn,
                    const float* bias, int bmode, unsigned bmod, int bstride,
                    int M, int N, int K, int relu, int nbatch) {
        dim3 grid((N + 63) / 64, (M + 127) / 128, nbatch);
        gemm_f16_wmma<<<grid, dim3(256), 0, stream>>>(A, Ab, Abmod, lda, Bm, Bb, sbk, sbn,
                                                      C, Cb, scm, scn, bias, bmode, bmod, bstride,
                                                      M, N, K, relu);
    };
    const int lnBlocks = (int)((Tok + 7) / 8);

    // ---- Block 1: self-attention over time -------------------------------
    gemm(x, 0, 1, Dc, Wq1, 0, 1, Dc, qb, 0, Dc, 1, bq1, 1, 1, 0, T, Dc, Dc, 0, 1);
    gemm(x, 0, 1, Dc, Wk1, 0, 1, Dc, kb, 0, Dc, 1, bk1, 1, 1, 0, T, Dc, Dc, 0, 1);
    gemm(x, 0, 1, Dc, Wv1, 0, 1, Dc, vb, 0, Dc, 1, bv1, 1, 1, 0, T, Dc, Dc, 0, 1);
    attn_time_wmma<<<Bc * Nc * Hc, 32, 0, stream>>>(qb, kb, vb, attn);
    gemm(attn, 0, 1, Dc, Wo1, 0, 1, Dc, proj, 0, Dc, 1, bo1, 1, 1, 0, T, Dc, Dc, 0, 1);
    ln_add_kernel<<<lnBlocks, 256, 0, stream>>>(x, proj, gx1, bx1, xln, Tok);

    // ---- Block 2: cross-attention (q from xln, k/v from memory) ----------
    gemm(xln,    0, 1, Dc, Wq2, 0, 1, Dc, qb, 0, Dc, 1, bq2, 1, 1, 0, T, Dc, Dc, 0, 1);
    gemm(memory, 0, 1, Dc, Wk2, 0, 1, Dc, kb, 0, Dc, 1, bk2, 1, 1, 0, T, Dc, Dc, 0, 1);
    gemm(memory, 0, 1, Dc, Wv2, 0, 1, Dc, vb, 0, Dc, 1, bv2, 1, 1, 0, T, Dc, Dc, 0, 1);
    attn_time_wmma<<<Bc * Nc * Hc, 32, 0, stream>>>(qb, kb, vb, attn);
    gemm(attn, 0, 1, Dc, Wo2, 0, 1, Dc, proj, 0, Dc, 1, bo2, 1, 1, 0, T, Dc, Dc, 0, 1);
    ln_add_kernel<<<lnBlocks, 256, 0, stream>>>(memory, proj, gm1, bm1, memln, Tok);

    // ---- FFN on x: out0 = LN(xln + relu(xln W1^T + b1) W2^T + b2) --------
    gemm(xln,   0, 1, Dc,   W1, 0, 1, Dc,   ffbuf, 0, DFFc, 1, b1, 1, 1, 0, T, DFFc, Dc,   1, 1);
    gemm(ffbuf, 0, 1, DFFc, W2, 0, 1, DFFc, ybuf,  0, Dc,   1, b2, 1, 1, 0, T, Dc,   DFFc, 0, 1);
    ln_add_kernel<<<lnBlocks, 256, 0, stream>>>(xln, ybuf, gx2, bx2, out0, Tok);

    // ---- FFN on memory: mff feeds As and m -------------------------------
    gemm(memln, 0, 1, Dc,   W3,  0, 1, Dc,   ffbuf, 0, DFFc, 1, b3,  1, 1, 0, T, DFFc, Dc,   1, 1);
    gemm(ffbuf, 0, 1, DFFc, WN2, 0, 1, DFFc, Asbuf, 0, Nc,   1, bN2, 1, 1, 0, T, Nc,   DFFc, 0, 1);
    gemm(ffbuf, 0, 1, DFFc, W4,  0, 1, DFFc, mbuf,  0, Dc,   1, b4,  1, 1, 0, T, Dc,   DFFc, 0, 1);
    ln_add_kernel<<<lnBlocks, 256, 0, stream>>>(mbuf, memln, gm2, bm2, out1, Tok);

    // ---- GCN: h = [x0, s1, s2, a1, a2], dts = Wg[l] @ h + bg[l] ----------
    transpose_x0<<<(int)((SZ_D + 255) / 256), 256, 0, stream>>>(data, hbuf, SZ_D);
    const unsigned long long HB = (unsigned long long)CINc * Nc;  // per-(b,l) h stride
    const int SL = Dc * Nc;                                       // slice stride within h
    const int BL = Bc * Lc;
    // s1 = x0 @ support ; s2 = s1 @ support
    gemm(hbuf,          HB, BIG, Nc, support, 0, Nc, 1, hbuf + SL,     HB, Nc, 1, nullptr, 0, 1, 0, Dc, Nc, Nc, 0, BL);
    gemm(hbuf + SL,     HB, BIG, Nc, support, 0, Nc, 1, hbuf + 2 * SL, HB, Nc, 1, nullptr, 0, 1, 0, Dc, Nc, Nc, 0, BL);
    // a1 = x0 @ As[b,l] ; a2 = a1 @ As[b,l]
    gemm(hbuf,          HB, BIG, Nc, Asbuf, (unsigned long long)Nc * Nc, Nc, 1, hbuf + 3 * SL, HB, Nc, 1, nullptr, 0, 1, 0, Dc, Nc, Nc, 0, BL);
    gemm(hbuf + 3 * SL, HB, BIG, Nc, Asbuf, (unsigned long long)Nc * Nc, Nc, 1, hbuf + 4 * SL, HB, Nc, 1, nullptr, 0, 1, 0, Dc, Nc, Nc, 0, BL);
    // dts[b,l,n,o] = sum_c Wg[l,o,c] h[b,l,c,n] + bg[l,o]   (C stored transposed via strides)
    gemm(Wg, (unsigned long long)Dc * CINc, (unsigned)Lc, CINc,
         hbuf, HB, Nc, 1,
         out2, (unsigned long long)Nc * Dc, 1, Dc,
         bg, 2, (unsigned)Lc, Dc,
         Dc, Nc, CINc, 0, BL);
}
